// ASReader_Modeler_50517405336212
// MI455X (gfx1250) — matrix operand order repaired and checked by reading, not compiled, not run on hardware
//
#include <hip/hip_runtime.h>

typedef _Float16 half_t;
typedef __attribute__((ext_vector_type(16))) _Float16 v16h;
typedef __attribute__((ext_vector_type(8)))  float    v8f;
typedef __attribute__((ext_vector_type(4)))  unsigned int v4u;
typedef __attribute__((ext_vector_type(8)))  int v8i;
typedef __attribute__((ext_vector_type(4)))  int v4i;

#define BB 64      // batch
#define TCC 512    // context length
#define TQQ 64     // question length
#define EE 300     // embed dim
#define HH 192     // hidden
#define H3 576     // 3*H
#define NCC 20     // candidates
#define KP 320     // E padded to multiple of 32
#define KT_IH 10   // KP/32 k-tiles for input projection
#define KT_HH 6    // HH/32 k-tiles for recurrent matmul
#define NN 1152    // 2 * 3H (fwd|bwd weights fused on N)
#define TILE_H 512 // halfs per swizzled 16x32(A)/32x16(B) tile: 32 lanes * 16

// ---------------------------------------------------------------------------
// All WMMA operands are PRE-SWIZZLED into per-lane-contiguous fragments:
// a tile is 32 lanes x 16 halfs (32B per lane) so operand loads are
// b128-pair vector loads instead of 16 strided u16 loads.
//
// A 16x32 f16 (ISA 7.12.2): lane = hf*16 + (m%16), hf = (kk>>3)&1,
//   element e = 2*v + (kk&1), v = (kk&16 ? 4:0) + ((kk&7)>>1)
// B 32x16 f16: lane = hf*16 + (n%16), element e = (kk&16 ? 8:0) + (kk&7)
// C/D 16x16 f32: VGPR r -> M = r + 8*(lane/16), N = lane%16
// ---------------------------------------------------------------------------
__device__ __forceinline__ v16h ld_v16h(const half_t* p) { return *(const v16h*)p; }

// ---------------------------------------------------------------------------
// Tensor Data Mover: contiguous/strided 2D tile of 2-byte data into LDS
// (D# bitfields per CDNA5 ISA).
// ---------------------------------------------------------------------------
__device__ __forceinline__ void tdm_load_tile_2d_f16(unsigned int lds_addr, const void* gptr,
                                                     unsigned int tile_d0, unsigned int tile_d1,
                                                     unsigned int tensor_d0, unsigned int tensor_d1,
                                                     unsigned int row_stride_elems) {
  const unsigned long long ga = (unsigned long long)(size_t)gptr;
  v4u g0;
  g0.x = 1u;                                        // count=1, user descriptor
  g0.y = lds_addr;                                  // D#.lds_addr
  g0.z = (unsigned int)ga;                          // global_addr[31:0]
  g0.w = (unsigned int)(ga >> 32) | (2u << 30);     // global_addr[56:32] | type=2
  v8i g1;
  g1[0] = (int)(1u << 16);                                              // data_size=1 (2B)
  g1[1] = (int)((tensor_d0 & 0xFFFFu) << 16);                           // tensor_dim0[15:0]
  g1[2] = (int)((tensor_d0 >> 16) | ((tensor_d1 & 0xFFFFu) << 16));     // dim0 hi | dim1 lo
  g1[3] = (int)((tensor_d1 >> 16) | (tile_d0 << 16));                   // dim1 hi | tile_dim0
  g1[4] = (int)(tile_d1 & 0xFFFFu);                                     // tile_dim1 (tile_dim2=0)
  g1[5] = (int)row_stride_elems;                                        // tensor_dim0_stride lo
  g1[6] = 0;
  g1[7] = 0;
  v4i z4 = {0, 0, 0, 0};
#if defined(__clang_major__) && (__clang_major__ >= 23)
  v8i z8 = {0, 0, 0, 0, 0, 0, 0, 0};
  __builtin_amdgcn_tensor_load_to_lds(g0, g1, z4, z4, z8, 0);
#else
  __builtin_amdgcn_tensor_load_to_lds(g0, g1, z4, z4, 0);
#endif
}

// ---------------------------------------------------------------------------
// Prep kernels
// ---------------------------------------------------------------------------
__global__ void zero_out_kernel(float* out) {
  if (threadIdx.x < 2) out[threadIdx.x] = 0.f;
}

// Embedding gather, written A-swizzled: X[((m/16)*KT_IH + k/32)*512 + lane*16 + e]
__global__ void gather_embed_kernel(const int* __restrict__ toks, const float* __restrict__ emb,
                                    half_t* __restrict__ X, int T, int total) {
  for (int idx = blockIdx.x * blockDim.x + threadIdx.x; idx < total; idx += gridDim.x * blockDim.x) {
    const int m = idx / KP, k = idx - m * KP;
    const int b = m & (BB - 1), t = m / BB;
    const int tok = toks[b * T + t];
    const float v = (k < EE) ? emb[(size_t)tok * EE + k] : 0.f;
    const int mt = m >> 4, mr = m & 15;
    const int kt = k >> 5, kk = k & 31;
    const int lane = (((kk >> 3) & 1) << 4) + mr;
    const int vv = ((kk & 16) ? 4 : 0) + ((kk & 7) >> 1);
    const int e = 2 * vv + (kk & 1);
    X[(((size_t)mt * KT_IH + kt) << 9) + lane * 16 + e] = (half_t)v;
  }
}

// Input weights, B-swizzled, strip-contiguous: Wt[[n/64][k/32][(n/16)&3][512]]
// so each 64-wide N-strip (40KB) is one contiguous TDM copy. bcat = [bA|bB].
__global__ void pack_wih_kernel(const float* __restrict__ WA, const float* __restrict__ WB,
                                const float* __restrict__ bA, const float* __restrict__ bB,
                                half_t* __restrict__ Wt, float* __restrict__ bcat) {
  const int start = blockIdx.x * blockDim.x + threadIdx.x;
  for (int idx = start; idx < KP * NN; idx += gridDim.x * blockDim.x) {
    const int k = idx / NN, n = idx - k * NN;
    float v = 0.f;
    if (k < EE) v = (n < H3) ? WA[n * EE + k] : WB[(n - H3) * EE + k];
    const int strip = n >> 6, nt = (n >> 4) & 3, nc = n & 15;
    const int kt = k >> 5, kk = k & 31;
    const int lane = (((kk >> 3) & 1) << 4) + nc;
    const int e = ((kk & 16) ? 8 : 0) + (kk & 7);
    Wt[((((size_t)strip * KT_IH + kt) * 4 + nt) << 9) + lane * 16 + e] = (half_t)v;
  }
  if (start < NN) bcat[start] = (start < H3) ? bA[start] : bB[start - H3];
}

// Recurrent weights, B-swizzled per scan: W[s][[n/16][k/32][512]]
__global__ void pack_whh_kernel(const float* __restrict__ w0, const float* __restrict__ w1,
                                const float* __restrict__ w2, const float* __restrict__ w3,
                                half_t* __restrict__ out) {
  const int per = HH * H3;
  for (int idx = blockIdx.x * blockDim.x + threadIdx.x; idx < 4 * per; idx += gridDim.x * blockDim.x) {
    const int s = idx / per, r = idx - s * per;
    const int k = r / H3, n = r - k * H3;
    const float* w = (s == 0) ? w0 : (s == 1) ? w1 : (s == 2) ? w2 : w3;
    const float v = w[n * HH + k];
    const int nt = n >> 4, nc = n & 15;
    const int kt = k >> 5, kk = k & 31;
    const int lane = (((kk >> 3) & 1) << 4) + nc;
    const int e = ((kk & 16) ? 8 : 0) + (kk & 7);
    out[(size_t)s * per + (((size_t)nt * KT_HH + kt) << 9) + lane * 16 + e] = (half_t)v;
  }
}

// ---------------------------------------------------------------------------
// GI = X @ Wih^T + bih. 256 thr = 8 waves, macrotile M=128 x N=64.
// Swizzled 40KB B-strip DMA'd to LDS by TDM; each wave owns one M-tile,
// sweeps 4 N-tiles, 10 WMMA k-steps per N-tile, A via b128 global loads.
// ---------------------------------------------------------------------------
__device__ __forceinline__ void store_c_tile(float* __restrict__ GI, int m0, int col0, int lane,
                                             v8f acc, const float* __restrict__ bias) {
  const int nc = lane & 15, mo = (lane >> 4) * 8;
  const int col = col0 + nc;
  const float bv = bias[col];
#pragma unroll
  for (int r = 0; r < 8; ++r)
    GI[(size_t)(m0 + mo + r) * NN + col] = acc[r] + bv;
}

__global__ void __launch_bounds__(256) gemm_gi_kernel(const half_t* __restrict__ X,
                                                      const half_t* __restrict__ Wt,
                                                      const float* __restrict__ bias,
                                                      float* __restrict__ GI) {
  __shared__ alignas(32) half_t Btile[KT_IH * 4 * TILE_H];  // 40KB swizzled B strip
  const int strip = blockIdx.y;
  const int m0b = blockIdx.x * 128;
  const int lane = threadIdx.x & 31;
  const int wave = threadIdx.x >> 5;

  if (threadIdx.x < 32) {  // wave 0 drives the TDM (1-D contiguous tile copy)
    const unsigned int lds_addr = (unsigned int)(size_t)(void*)&Btile[0];
    const unsigned int elems = KT_IH * 4 * TILE_H;  // 20480
    tdm_load_tile_2d_f16(lds_addr, Wt + (size_t)strip * elems, elems, 1u, elems, 1u, elems);
    __builtin_amdgcn_s_wait_tensorcnt(0);
  }
  __syncthreads();

  const int m0 = m0b + wave * 16;
  const half_t* Ap = X + (((size_t)(m0 >> 4)) * KT_IH << 9) + lane * 16;
  const half_t* Bp = Btile + lane * 16;
  v8f a0 = {}, a1 = {}, a2 = {}, a3 = {};
  for (int kt = 0; kt < KT_IH; ++kt) {
    __builtin_prefetch(Ap + TILE_H, 0, 1);
    const v16h a = ld_v16h(Ap);
    const v16h b0 = ld_v16h(Bp + ((kt * 4 + 0) << 9));
    a0 = __builtin_amdgcn_wmma_f32_16x16x32_f16(false, a, false, b0, (short)0, a0, false, false);
    const v16h b1 = ld_v16h(Bp + ((kt * 4 + 1) << 9));
    a1 = __builtin_amdgcn_wmma_f32_16x16x32_f16(false, a, false, b1, (short)0, a1, false, false);
    const v16h b2 = ld_v16h(Bp + ((kt * 4 + 2) << 9));
    a2 = __builtin_amdgcn_wmma_f32_16x16x32_f16(false, a, false, b2, (short)0, a2, false, false);
    const v16h b3 = ld_v16h(Bp + ((kt * 4 + 3) << 9));
    a3 = __builtin_amdgcn_wmma_f32_16x16x32_f16(false, a, false, b3, (short)0, a3, false, false);
    Ap += TILE_H;
  }
  const int n0s = strip * 64;
  store_c_tile(GI, m0, n0s + 0, lane, a0, bias);
  store_c_tile(GI, m0, n0s + 16, lane, a1, bias);
  store_c_tile(GI, m0, n0s + 32, lane, a2, bias);
  store_c_tile(GI, m0, n0s + 48, lane, a3, bias);
}

// ---------------------------------------------------------------------------
// Recurrent GRU scans. 4 blocks (cf, cb, qf, qb), 256 threads = 8 waves.
// Per step: gh = h @ Whh^T + bhh via WMMA (144 tiles / 8 waves, K=192),
// then fused gate math. h kept in LDS: f32 row-major + f16 A-SWIZZLED
// ([kt][mt][512]) so phase-1 A fragments are vector DS loads.
// Backward-question scan only needs its first step (reference uses qb_out[0]).
// ---------------------------------------------------------------------------
__global__ void __launch_bounds__(256) gru_scan_kernel(
    const float* __restrict__ GIc, const float* __restrict__ GIq,
    const half_t* __restrict__ WhhT,
    const float* __restrict__ bhh_cf, const float* __restrict__ bhh_cb,
    const float* __restrict__ bhh_qf, const float* __restrict__ bhh_qb,
    const float* __restrict__ h0_cf, const float* __restrict__ h0_cb,
    const float* __restrict__ h0_qf, const float* __restrict__ h0_qb,
    float* __restrict__ outf, float* __restrict__ outb, float* __restrict__ qlast) {
  extern __shared__ char smem[];
  float*  gh  = (float*)smem;                                    // 64*576 f32 = 144KB
  float*  h32 = (float*)(smem + BB * H3 * 4);                    // 64*192 f32 = 48KB
  half_t* h16 = (half_t*)(smem + BB * H3 * 4 + BB * HH * 4);     // 64*192 f16 = 24KB (A-swizzled)

  const int s = blockIdx.x;
  const float* GI   = (s < 2) ? GIc : GIq;
  const int    T    = (s < 2) ? TCC : TQQ;
  const int    dir  = (s & 1) ? -1 : 1;
  const int    ncoff = (s & 1) ? H3 : 0;
  const float* bhh = (s == 0) ? bhh_cf : (s == 1) ? bhh_cb : (s == 2) ? bhh_qf : bhh_qb;
  const float* h0  = (s == 0) ? h0_cf : (s == 1) ? h0_cb : (s == 2) ? h0_qf : h0_qb;
  const half_t* W  = WhhT + (size_t)s * HH * H3;
  float* oseq = (s == 0) ? outf : (s == 1) ? outb : nullptr;
  const int steps = (s == 3) ? 1 : T;
  const int tid = threadIdx.x, lane = tid & 31, wave = tid >> 5;

  for (int i = tid; i < BB * HH; i += 256) {
    const int b = i / HH, j = i - b * HH;
    const float v = h0[i];
    h32[i] = v;
    const int kt = j >> 5, kk = j & 31;
    const int lane2 = (((kk >> 3) & 1) << 4) + (b & 15);
    const int vv = ((kk & 16) ? 4 : 0) + ((kk & 7) >> 1);
    h16[(((kt << 2) + (b >> 4)) << 9) + lane2 * 16 + 2 * vv + (kk & 1)] = (half_t)v;
  }
  __syncthreads();

  for (int it = 0; it < steps; ++it) {
    const int t = (dir > 0) ? it : (T - 1 - it);
    // phase 1: gh[64][576] = h16 @ W + bhh   (uniform control flow -> EXEC all-1s for WMMA)
    for (int tile = wave; tile < 144; tile += 8) {
      const int mt = tile & 3, nt = tile >> 2;
      const half_t* Ap = h16 + ((size_t)mt << 9) + lane * 16;                  // [kt][mt][512]
      const half_t* Bp = W + (((size_t)nt * KT_HH) << 9) + lane * 16;          // [nt][kt][512]
      v8f acc = {};
#pragma unroll
      for (int kt = 0; kt < KT_HH; ++kt) {
        const v16h a = ld_v16h(Ap + ((size_t)kt << 11));   // kt*4*512
        const v16h b = ld_v16h(Bp + ((size_t)kt << 9));
        acc = __builtin_amdgcn_wmma_f32_16x16x32_f16(false, a, false, b, (short)0, acc, false, false);
      }
      const int nc = lane & 15, mo = (lane >> 4) * 8;
      const int n0 = nt * 16;
      const float bv = bhh[n0 + nc];
#pragma unroll
      for (int r = 0; r < 8; ++r) gh[(mt * 16 + mo + r) * H3 + n0 + nc] = acc[r] + bv;
    }
    __syncthreads();
    // phase 2: gate math, update h (both copies), emit outputs
    for (int i = tid; i < BB * HH; i += 256) {
      const int b = i / HH, j = i - b * HH;
      const float* gi = GI + ((size_t)t * BB + b) * NN + ncoff;
      const float* g = gh + b * H3;
      const float r = 1.f / (1.f + __expf(-(gi[j] + g[j])));
      const float z = 1.f / (1.f + __expf(-(gi[HH + j] + g[HH + j])));
      const float n = tanhf(gi[2 * HH + j] + r * g[2 * HH + j]);
      const float hn = tanhf((1.f - z) * n + z * h32[i]);
      h32[i] = hn;
      const int kt = j >> 5, kk = j & 31;
      const int lane2 = (((kk >> 3) & 1) << 4) + (b & 15);
      const int vv = ((kk & 16) ? 4 : 0) + ((kk & 7) >> 1);
      h16[(((kt << 2) + (b >> 4)) << 9) + lane2 * 16 + 2 * vv + (kk & 1)] = (half_t)hn;
      if (oseq) oseq[((size_t)t * BB + b) * HH + j] = hn;
      if (s == 2 && it == T - 1) qlast[b * (2 * HH) + j] = hn;        // qf last state
      if (s == 3 && it == 0) qlast[b * (2 * HH) + HH + j] = hn;       // qb first step = qb_out[0]
    }
    __syncthreads();
  }
}

// ---------------------------------------------------------------------------
// Attention + candidate scoring + log-softmax + loss/acc. One block per b.
// ---------------------------------------------------------------------------
__global__ void __launch_bounds__(256) score_kernel(const int* __restrict__ context,
                                                    const int* __restrict__ candidates,
                                                    const int* __restrict__ target,
                                                    const float* __restrict__ outf,
                                                    const float* __restrict__ outb,
                                                    const float* __restrict__ qlast,
                                                    float* __restrict__ out) {
  __shared__ float ql[2 * HH];
  __shared__ float dp[TCC];
  __shared__ float sc[NCC];
  const int b = blockIdx.x, tid = threadIdx.x;

  for (int i = tid; i < 2 * HH; i += 256) ql[i] = qlast[b * (2 * HH) + i];
  __syncthreads();

  for (int t = tid; t < TCC; t += 256) {
    float d = 0.f;
    const float* pf = outf + ((size_t)t * BB + b) * HH;
    const float* pb = outb + ((size_t)t * BB + b) * HH;
    for (int j = 0; j < HH; ++j) d += pf[j] * ql[j] + pb[j] * ql[HH + j];
    dp[t] = (context[b * TCC + t] != 0) ? d : 0.f;
  }
  __syncthreads();

  if (tid < NCC) {
    const int cand = candidates[b * NCC + tid];
    float S = 0.f;
    for (int t = 0; t < TCC; ++t)
      if (context[b * TCC + t] == cand) S += dp[t];
    int cnt = 0;
    for (int j = 0; j < NCC; ++j) {
      const int cj = candidates[b * NCC + j];
      cnt += (cand == cj && cj != 0) ? 1 : 0;
    }
    sc[tid] = S * (float)cnt;
  }
  __syncthreads();

  if (tid == 0) {
    float mx = sc[0];
    for (int i = 1; i < NCC; ++i) mx = fmaxf(mx, sc[i]);
    float se = 0.f;
    for (int i = 0; i < NCC; ++i) se += __expf(sc[i] - mx);
    const float lse = mx + __logf(se);
    int amax = 0;
    float best = sc[0];
    for (int i = 0; i < NCC; ++i) {
      out[2 + b * NCC + i] = sc[i] - lse;
      if (sc[i] > best) { best = sc[i]; amax = i; }
    }
    const int tg = target[b];
    atomicAdd(&out[0], -(sc[tg] - lse) * (1.f / (float)BB));
    atomicAdd(&out[1], (amax == tg) ? (1.f / (float)BB) : 0.f);
  }
}

// ---------------------------------------------------------------------------
extern "C" void kernel_launch(void* const* d_in, const int* in_sizes, int n_in,
                              void* d_out, int out_size, void* d_ws, size_t ws_size,
                              hipStream_t stream) {
  (void)in_sizes; (void)n_in; (void)out_size; (void)ws_size;
  const int* context    = (const int*)d_in[0];
  const int* question   = (const int*)d_in[1];
  const int* candidates = (const int*)d_in[2];
  const int* target     = (const int*)d_in[3];
  const float* emb      = (const float*)d_in[4];
  const float* cf_Wih = (const float*)d_in[5],  *cf_Whh = (const float*)d_in[6];
  const float* cf_bih = (const float*)d_in[7],  *cf_bhh = (const float*)d_in[8];
  const float* cb_Wih = (const float*)d_in[9],  *cb_Whh = (const float*)d_in[10];
  const float* cb_bih = (const float*)d_in[11], *cb_bhh = (const float*)d_in[12];
  const float* qf_Wih = (const float*)d_in[13], *qf_Whh = (const float*)d_in[14];
  const float* qf_bih = (const float*)d_in[15], *qf_bhh = (const float*)d_in[16];
  const float* qb_Wih = (const float*)d_in[17], *qb_Whh = (const float*)d_in[18];
  const float* qb_bih = (const float*)d_in[19], *qb_bhh = (const float*)d_in[20];
  const float* h0_cf = (const float*)d_in[21], *h0_cb = (const float*)d_in[22];
  const float* h0_qf = (const float*)d_in[23], *h0_qb = (const float*)d_in[24];

  char* ws = (char*)d_ws;
  size_t off = 0;
  auto alloc = [&](size_t bytes) -> void* {
    void* p = ws + off;
    off = (off + bytes + 255) & ~(size_t)255;
    return p;
  };
  half_t* Xc  = (half_t*)alloc((size_t)BB * TCC * KP * 2);   // 21.0 MB (A-swizzled)
  half_t* Xq  = (half_t*)alloc((size_t)BB * TQQ * KP * 2);   //  2.6 MB (A-swizzled)
  half_t* WtC = (half_t*)alloc((size_t)KP * NN * 2);         //  0.7 MB (B-swizzled)
  half_t* WtQ = (half_t*)alloc((size_t)KP * NN * 2);         //  0.7 MB (B-swizzled)
  half_t* Whh = (half_t*)alloc((size_t)4 * HH * H3 * 2);     //  0.9 MB (B-swizzled)
  float*  biC = (float*)alloc((size_t)NN * 4);
  float*  biQ = (float*)alloc((size_t)NN * 4);
  float*  GIc = (float*)alloc((size_t)BB * TCC * NN * 4);    // 151 MB
  float*  GIq = (float*)alloc((size_t)BB * TQQ * NN * 4);    //  19 MB
  float*  outF = (float*)alloc((size_t)TCC * BB * HH * 4);   //  25 MB
  float*  outB = (float*)alloc((size_t)TCC * BB * HH * 4);   //  25 MB
  float*  qlast = (float*)alloc((size_t)BB * 2 * HH * 4);

  zero_out_kernel<<<1, 32, 0, stream>>>((float*)d_out);

  gather_embed_kernel<<<4096, 256, 0, stream>>>(context, emb, Xc, TCC, BB * TCC * KP);
  gather_embed_kernel<<<1024, 256, 0, stream>>>(question, emb, Xq, TQQ, BB * TQQ * KP);
  pack_wih_kernel<<<1440, 256, 0, stream>>>(cf_Wih, cb_Wih, cf_bih, cb_bih, WtC, biC);
  pack_wih_kernel<<<1440, 256, 0, stream>>>(qf_Wih, qb_Wih, qf_bih, qb_bih, WtQ, biQ);
  pack_whh_kernel<<<1728, 256, 0, stream>>>(cf_Whh, cb_Whh, qf_Whh, qb_Whh, Whh);

  gemm_gi_kernel<<<dim3((BB * TCC) / 128, NN / 64), 256, 0, stream>>>(Xc, WtC, biC, GIc);
  gemm_gi_kernel<<<dim3((BB * TQQ) / 128, NN / 64), 256, 0, stream>>>(Xq, WtQ, biQ, GIq);

  const size_t gru_lds = (size_t)BB * H3 * 4 + (size_t)BB * HH * 4 + (size_t)BB * HH * 2;  // 216KB < 320KB WGP LDS
  gru_scan_kernel<<<4, 256, gru_lds, stream>>>(GIc, GIq, Whh,
                                               cf_bhh, cb_bhh, qf_bhh, qb_bhh,
                                               h0_cf, h0_cb, h0_qf, h0_qb,
                                               outF, outB, qlast);

  score_kernel<<<BB, 256, 0, stream>>>(context, candidates, target, outF, outB, qlast, (float*)d_out);
}
